// BlockBERTlucidrains_49778670961053
// MI455X (gfx1250) — matrix operand order repaired and checked by reading, not compiled, not run on hardware
//
#include <hip/hip_runtime.h>
#include <hip/hip_fp16.h>

// ---------------------------------------------------------------------------
// BlockBERT forward for MI455X (gfx1250, wave32, WMMA, TDM).
// f16 WMMA GEMMs (fp32 accum), weights pre-swizzled to fragment order,
// A tiles double-buffered through LDS via tensor_load_to_lds (TENSORcnt).
// ---------------------------------------------------------------------------

typedef __attribute__((ext_vector_type(16))) _Float16 v16h;
typedef __attribute__((ext_vector_type(8)))  float    v8f;
typedef __attribute__((ext_vector_type(4)))  unsigned int u32x4;
typedef __attribute__((ext_vector_type(8)))  int      i32x8;
typedef __attribute__((ext_vector_type(4)))  int      i32x4;

union F16x16 { v16h v; _Float16 h[16]; uint4 u[2]; };
union F32x8  { v8f  v; float    f[8];  };

constexpr int Bb  = 32;
constexpr int Nn  = 512;
constexpr int Dd  = 512;
constexpr int Ll  = 4;
constexpr int Hh  = 8;
constexpr int HD  = 64;
constexpr int RECL= 2;
constexpr int MBr = Bb * Nn;   // 16384 token rows
constexpr int BH  = Bb * Hh;   // 256 (b,h) pairs
constexpr float SCALEF = 0.125f; // HD^-0.5

#define WMMA_F16(a,b,c) \
  __builtin_amdgcn_wmma_f32_16x16x32_f16(false,(a),false,(b),(short)0,(c),false,false)

constexpr int GF_GELU   = 1;
constexpr int GF_F16OUT = 2;

#if defined(__has_builtin)
#  if __has_builtin(__builtin_amdgcn_tensor_load_to_lds) && \
      __has_builtin(__builtin_amdgcn_s_wait_tensorcnt)
#    define HAVE_TDM 1
#  endif
#endif

#ifdef HAVE_TDM
// Issue a TDM load of a [rows x 32] f16 tile (row stride = K elements) into LDS.
// 2-D descriptor: groups 2/3 zero.  OOB rows/cols return zero per ISA 8.2.
__device__ __forceinline__ void tdm_load_a(unsigned int lds_off, unsigned long long gaddr,
                                           int rows_rem, int k_rem, int K) {
  u32x4 g0;
  g0[0] = 1u;                                           // count=1, user mode
  g0[1] = lds_off;                                      // LDS byte address
  g0[2] = (unsigned int)gaddr;                          // global_addr[31:0]
  g0[3] = ((unsigned int)(gaddr >> 32) & 0x1FFFFFFu) | (2u << 30); // addr[56:32] | type=2
  i32x8 g1;
  g1[0] = 0x00010000;                                   // data_size=2B, mask=0
  g1[1] = (int)(((unsigned int)k_rem & 0xFFFFu) << 16); // tensor_dim0[15:0] @63:48
  g1[2] = (int)((((unsigned int)k_rem) >> 16) |         // tensor_dim0[31:16]
                (((unsigned int)rows_rem & 0xFFFFu) << 16)); // tensor_dim1[15:0]
  g1[3] = (int)((((unsigned int)rows_rem) >> 16) |      // tensor_dim1[31:16]
                (32u << 16));                           // tile_dim0 = 32
  g1[4] = 64;                                           // tile_dim1 = 64, tile_dim2 = 0
  g1[5] = K;                                            // tensor_dim0_stride low32
  g1[6] = 0;
  g1[7] = 0;
  i32x4 g2 = {0, 0, 0, 0};
  i32x4 g3 = {0, 0, 0, 0};
#if defined(__clang_major__) && (__clang_major__ >= 23)
  i32x8 g4 = {0, 0, 0, 0, 0, 0, 0, 0};
  __builtin_amdgcn_tensor_load_to_lds(g0, g1, g2, g3, g4, 0);
#else
  __builtin_amdgcn_tensor_load_to_lds(g0, g1, g2, g3, 0);
#endif
}
#endif

// ---------------------------------------------------------------------------
// Weight swizzle: B[K,N] fp32 row-major -> f16 WMMA B-fragment order.
// Tile (kc,nt) = 32 lanes x 16 f16 contiguous; lane L elem e =
//   B[kc*32 + (L>=16?16:0) + e][nt*16 + (L&15)]
// ---------------------------------------------------------------------------
__global__ __launch_bounds__(256)
void swizzle_b_kernel(const float* __restrict__ B, _Float16* __restrict__ Bsw,
                      int K, int N) {
  int idx   = blockIdx.x * 256 + threadIdx.x;   // over (K/32)*(N/16)*32
  int lane  = idx & 31;
  int tile  = idx >> 5;
  int ntile = N >> 4;
  int kc = tile / ntile, nt = tile - kc * ntile;
  int kbase = kc * 32 + (lane >> 4) * 16;
  int col   = nt * 16 + (lane & 15);
  F16x16 tmp;
#pragma unroll
  for (int e = 0; e < 16; e++)
    tmp.h[e] = (_Float16)B[(size_t)(kbase + e) * N + col];
  _Float16* dst = Bsw + ((size_t)tile * 32 + lane) * 16;
  *(uint4*)dst       = tmp.u[0];
  *(uint4*)(dst + 8) = tmp.u[1];
}

// ---------------------------------------------------------------------------
// token embedding gather + state copy
// ---------------------------------------------------------------------------
__global__ __launch_bounds__(256)
void embed_kernel(const int* __restrict__ ids, const float* __restrict__ emb,
                  float* __restrict__ x) {
  int i  = blockIdx.x * 256 + threadIdx.x;
  int d  = i & (Dd - 1);
  int bn = i >> 9;
  x[i] = emb[(size_t)ids[bn] * Dd + d];
}

__global__ __launch_bounds__(256)
void copy_state_kernel(const float* __restrict__ sin, float* __restrict__ s) {
  int i = blockIdx.x * 256 + threadIdx.x;
  s[i] = sin[i];
}

// ---------------------------------------------------------------------------
// LayerNorm over D=512; one block per row; f16 output
// ---------------------------------------------------------------------------
__global__ __launch_bounds__(256)
void ln_kernel(const float* __restrict__ x, const float* __restrict__ g,
               const float* __restrict__ bp, _Float16* __restrict__ out) {
  __shared__ float red[256];
  int row = blockIdx.x, t = threadIdx.x;
  const float* xr = x + (size_t)row * Dd;
  float a0 = xr[t], a1 = xr[t + 256];
  red[t] = a0 + a1; __syncthreads();
  for (int o = 128; o > 0; o >>= 1) { if (t < o) red[t] += red[t + o]; __syncthreads(); }
  float mean = red[0] * (1.0f / Dd);
  __syncthreads();
  float d0 = a0 - mean, d1 = a1 - mean;
  red[t] = d0 * d0 + d1 * d1; __syncthreads();
  for (int o = 128; o > 0; o >>= 1) { if (t < o) red[t] += red[t + o]; __syncthreads(); }
  float rstd = rsqrtf(red[0] * (1.0f / Dd) + 1e-5f);
  out[(size_t)row * Dd + t]       = (_Float16)(d0 * rstd * g[t]       + bp[t]);
  out[(size_t)row * Dd + t + 256] = (_Float16)(d1 * rstd * g[t + 256] + bp[t + 256]);
}

// ---------------------------------------------------------------------------
// WMMA GEMM: C[M,N] = A[M,K] (f16 row-major) @ Bsw (f16, fragment order)
// Block 256 thr = 8 waves (2M x 4N), block tile 64 x 128, wave tile 32 x 32.
// A tiles double-buffered in LDS via TDM (issue next, compute current, wait
// TENSORcnt after the WMMAs).  B frags straight from global (L2-resident).
// Epilogue: +bias, gelu(tanh), +resid; f32 or f16 out.
// ---------------------------------------------------------------------------
__global__ __launch_bounds__(256)
void gemm_kernel(const _Float16* __restrict__ A, const _Float16* __restrict__ Bsw,
                 const float* __restrict__ bias, const float* __restrict__ resid,
                 float* __restrict__ Cf, _Float16* __restrict__ Ch,
                 int M, int N, int K, int flags) {
#ifdef HAVE_TDM
  __shared__ _Float16 As[2][64][32];
#else
  __shared__ _Float16 As[1][64][32];
#endif
  int n0 = blockIdx.x * 128;
  int m0 = blockIdx.y * 64;
  int t = threadIdx.x;
  int lane = t & 31, wave = t >> 5;
  int wm = wave & 1, wn = wave >> 1;           // 2(M) x 4(N) waves
  int r = lane & 15, hi = lane >> 4;
  int kb = hi * 16;
  int ntile = N >> 4;
  int nk = K >> 5;

  F32x8 acc[2][2];
#pragma unroll
  for (int i2 = 0; i2 < 2; i2++)
#pragma unroll
    for (int j2 = 0; j2 < 2; j2++)
#pragma unroll
      for (int i = 0; i < 8; i++) acc[i2][j2].f[i] = 0.0f;

#ifdef HAVE_TDM
  unsigned int lds0 = (unsigned int)(size_t)&As[0][0][0];
  unsigned int lds1 = (unsigned int)(size_t)&As[1][0][0];
  // prologue: chunk 0 -> buffer 0
  if (wave == 0) {
    tdm_load_a(lds0, (unsigned long long)(size_t)(A + (size_t)m0 * K),
               M - m0, K, K);
    __builtin_amdgcn_s_wait_tensorcnt(0);
  }
  __syncthreads();
#else
  int arow = t >> 2, acol = (t & 3) * 8;       // manual A stage: 8 f16/thread
#endif

  for (int kc = 0; kc < nk; ++kc) {
    int cur = kc & 1;
#ifdef HAVE_TDM
    // issue DMA for the NEXT chunk into the idle buffer before computing
    if (wave == 0 && kc + 1 < nk) {
      int k1 = (kc + 1) * 32;
      tdm_load_a(cur ? lds0 : lds1,
                 (unsigned long long)(size_t)(A + (size_t)m0 * K + k1),
                 M - m0, K - k1, K);
    }
#else
    int k0 = kc * 32;
    if (m0 + arow < M)
      *(uint4*)&As[0][arow][acol] =
          *(const uint4*)(A + (size_t)(m0 + arow) * K + k0 + acol);
    else
      *(uint4*)&As[0][arow][acol] = make_uint4(0u, 0u, 0u, 0u);
    if (k0 + 32 < K && m0 + arow < M)
      __builtin_prefetch(A + (size_t)(m0 + arow) * K + k0 + 32 + acol, 0, 3);
    __syncthreads();
    cur = 0;
#endif

    F16x16 af[2], bf[2];
#pragma unroll
    for (int i2 = 0; i2 < 2; i2++) {
      int row = wm * 32 + i2 * 16 + r;
      af[i2].u[0] = *(const uint4*)&As[cur][row][kb];
      af[i2].u[1] = *(const uint4*)&As[cur][row][kb + 8];
    }
#pragma unroll
    for (int j2 = 0; j2 < 2; j2++) {
      int nt = blockIdx.x * 8 + wn * 2 + j2;
      const _Float16* bp = Bsw + (((size_t)kc * ntile + nt) * 32 + lane) * 16;
      bf[j2].u[0] = *(const uint4*)bp;
      bf[j2].u[1] = *(const uint4*)(bp + 8);
    }
#pragma unroll
    for (int i2 = 0; i2 < 2; i2++)
#pragma unroll
      for (int j2 = 0; j2 < 2; j2++)
        acc[i2][j2].v = WMMA_F16(af[i2].v, bf[j2].v, acc[i2][j2].v);

#ifdef HAVE_TDM
    // DMA for next chunk has been in flight behind the WMMAs; retire it now
    if (wave == 0 && kc + 1 < nk) __builtin_amdgcn_s_wait_tensorcnt(0);
#endif
    __syncthreads();
  }

#pragma unroll
  for (int i2 = 0; i2 < 2; i2++)
#pragma unroll
    for (int j2 = 0; j2 < 2; j2++) {
      int coln = n0 + (wn * 2 + j2) * 16 + r;
#pragma unroll
      for (int i = 0; i < 8; i++) {
        int row = m0 + wm * 32 + i2 * 16 + i + 8 * hi;
        if (row >= M) continue;
        float v = acc[i2][j2].f[i];
        if (bias) v += bias[coln];
        if (flags & GF_GELU) {
          float u = v;
          v = 0.5f * u * (1.0f + tanhf(0.7978845608f * (u + 0.044715f * u * u * u)));
        }
        if (resid) v += resid[(size_t)row * N + coln];
        if (flags & GF_F16OUT) Ch[(size_t)row * N + coln] = (_Float16)v;
        else                   Cf[(size_t)row * N + coln] = v;
      }
    }
}

// ---------------------------------------------------------------------------
// RoPE on q,k + split heads. q,k -> (B*H, N, HD) row-major;
// v -> TRANSPOSED (B*H, HD, N) so attention P@V B-frags are contiguous.
// ---------------------------------------------------------------------------
__global__ __launch_bounds__(256)
void rope_split_kernel(const _Float16* __restrict__ qkv,
                       _Float16* __restrict__ q, _Float16* __restrict__ k,
                       _Float16* __restrict__ vt) {
  int i = blockIdx.x * 256 + threadIdx.x;      // over Bb*Nn*Hh*32
  int p = i & 31; int tmp = i >> 5;
  int h = tmp & 7; tmp >>= 3;
  int n = tmp & 511; int b = tmp >> 9;
  size_t row = (size_t)(b * Nn + n) * (3 * Dd);
  float inv = __expf(-(2.0f * p / (float)HD) * 9.210340372f); // 10000^(-2p/HD)
  float ang = (float)n * inv;
  float cs = __cosf(ang), sn = __sinf(ang);
  int qc = h * HD;
  float q1 = (float)qkv[row + qc + p],      q2 = (float)qkv[row + qc + 32 + p];
  float k1 = (float)qkv[row + Dd + qc + p], k2 = (float)qkv[row + Dd + qc + 32 + p];
  size_t orow = ((size_t)(b * Hh + h) * Nn + n) * HD;
  q[orow + p]      = (_Float16)(q1 * cs - q2 * sn);
  q[orow + 32 + p] = (_Float16)(q2 * cs + q1 * sn);
  k[orow + p]      = (_Float16)(k1 * cs - k2 * sn);
  k[orow + 32 + p] = (_Float16)(k2 * cs + k1 * sn);
  size_t vrow = (size_t)(b * Hh + h) * HD;
  vt[(vrow + p)      * Nn + n] = qkv[row + 2 * Dd + qc + p];
  vt[(vrow + 32 + p) * Nn + n] = qkv[row + 2 * Dd + qc + 32 + p];
}

// split state qkv (no rope): sqkv (Bb x 1536) -> sq/sk/sv (B*H, HD)
__global__ __launch_bounds__(256)
void split_state_kernel(const _Float16* __restrict__ sqkv,
                        _Float16* __restrict__ sq, _Float16* __restrict__ sk,
                        _Float16* __restrict__ sv) {
  int i = blockIdx.x * 256 + threadIdx.x;      // over BH*HD = 16384
  int d = i & (HD - 1);
  int h = (i >> 6) & 7;
  int b = i >> 9;
  size_t rb = (size_t)b * (3 * Dd);
  sq[i] = sqkv[rb +          h * HD + d];
  sk[i] = sqkv[rb + Dd     + h * HD + d];
  sv[i] = sqkv[rb + 2 * Dd + h * HD + d];
}

// ---------------------------------------------------------------------------
// Fused causal attention, one wave per ((b*H+h), 16-row tile).
// QK^T via WMMA (vectorized global frag loads), softmax -> f16 LDS tile,
// P@V via WMMA from transposed V.  Optional state column 512.
// ---------------------------------------------------------------------------
__global__ __launch_bounds__(32)
void attn_kernel(const _Float16* __restrict__ Qg, const _Float16* __restrict__ Kg,
                 const _Float16* __restrict__ Vt,
                 const _Float16* __restrict__ skg, const _Float16* __restrict__ svg,
                 _Float16* __restrict__ outg, int hasState) {
  __shared__ float    S[16][544];
  __shared__ _Float16 P16[16][544];
  int bh = blockIdx.x, rt = blockIdx.y;
  int b = bh >> 3, h = bh & 7;
  int lane = threadIdx.x;
  int r = lane & 15, hi = lane >> 4;
  int kb = hi * 8, kb2 = hi * 16;

  const _Float16* Qp = Qg + ((size_t)bh * Nn + rt * 16) * HD;
  const _Float16* Kp = Kg + (size_t)bh * Nn * HD;
  const _Float16* Vp = Vt + (size_t)bh * HD * Nn;   // (HD, Nn)

  // Q fragments (16x64 = two 16x32 chunks), contiguous b128 loads
  F16x16 qa[2];
#pragma unroll
  for (int c2 = 0; c2 < 2; c2++) {
    const _Float16* qp = Qp + r * HD + c2 * 32 + kb2;
    qa[c2].u[0] = *(const uint4*)qp;
    qa[c2].u[1] = *(const uint4*)(qp + 8);
  }

  int nkeys = rt * 16 + 16;
  int npad  = (nkeys + 31) & ~31;

  for (int j = 0; j <= rt; ++j) {
    F32x8 c;
#pragma unroll
    for (int i = 0; i < 8; i++) c.f[i] = 0.0f;
    int keyc = j * 16 + r;                       // B-frag column = key index
#pragma unroll
    for (int c2 = 0; c2 < 2; c2++) {
      F16x16 bfr;
      const _Float16* kp = Kp + (size_t)keyc * HD + c2 * 32 + kb2;
      bfr.u[0] = *(const uint4*)kp;
      bfr.u[1] = *(const uint4*)(kp + 8);
      c.v = WMMA_F16(qa[c2].v, bfr.v, c.v);
    }
#pragma unroll
    for (int i = 0; i < 8; i++) {
      int lr = i + 8 * hi;
      int rowg = rt * 16 + lr;
      int colg = j * 16 + r;
      float sv2 = c.f[i] * SCALEF;
      S[lr][colg] = (colg <= rowg) ? sv2 : -1e9f;
    }
  }

  if (hasState && lane < 16) {                   // state key score, column 512
    const _Float16* skp = skg + (size_t)bh * HD;
    float a2 = 0.0f;
    for (int d = 0; d < HD; d++)
      a2 += (float)Qp[lane * HD + d] * (float)skp[d];
    S[lane][512] = a2 * SCALEF;
  }
  __syncthreads();

  // per-row softmax (lanes 0..15); probabilities written as f16 into P16
  if (lane < 16) {
    float mx = -1e30f;
    for (int jc = 0; jc < nkeys; jc++) mx = fmaxf(mx, S[lane][jc]);
    if (hasState) mx = fmaxf(mx, S[lane][512]);
    float ssum = 0.0f;
    for (int jc = 0; jc < nkeys; jc++) ssum += __expf(S[lane][jc] - mx);
    float es = hasState ? __expf(S[lane][512] - mx) : 0.0f;
    float inv = 1.0f / (ssum + es);
    for (int jc = 0; jc < nkeys; jc++)
      P16[lane][jc] = (_Float16)(__expf(S[lane][jc] - mx) * inv);
    for (int jc = nkeys; jc < npad; jc++) P16[lane][jc] = (_Float16)0.0f;
    P16[lane][512] = (_Float16)(es * inv);
  }
  __syncthreads();

  // O(16x64) = P(16 x npad) @ V(npad x 64); 4 accumulators over HD
  F32x8 o[4];
#pragma unroll
  for (int nt = 0; nt < 4; nt++)
#pragma unroll
    for (int i = 0; i < 8; i++) o[nt].f[i] = 0.0f;

  for (int kc = 0; kc < npad; kc += 32) {
    F16x16 pf;
    pf.u[0] = *(const uint4*)&P16[r][kc + kb2];
    pf.u[1] = *(const uint4*)&P16[r][kc + kb2 + 8];
#pragma unroll
    for (int nt = 0; nt < 4; nt++) {
      F16x16 vf;
      const _Float16* vp = Vp + (size_t)(nt * 16 + r) * Nn + kc + kb2;
      vf.u[0] = *(const uint4*)vp;
      vf.u[1] = *(const uint4*)(vp + 8);
      o[nt].v = WMMA_F16(pf.v, vf.v, o[nt].v);
    }
  }
  if (hasState) {                                // rank-1 state-value term
    const _Float16* svp = svg + (size_t)bh * HD;
#pragma unroll
    for (int nt = 0; nt < 4; nt++)
#pragma unroll
      for (int i = 0; i < 8; i++) {
        int lr = i + 8 * hi;
        o[nt].f[i] += (float)P16[lr][512] * (float)svp[nt * 16 + r];
      }
  }
#pragma unroll
  for (int nt = 0; nt < 4; nt++)
#pragma unroll
    for (int i = 0; i < 8; i++) {
      int row = rt * 16 + i + 8 * hi;
      outg[((size_t)(b * Nn + row)) * Dd + h * HD + nt * 16 + r] =
          (_Float16)o[nt].f[i];
    }
}

// ---------------------------------------------------------------------------
// state attention: 1 query per (b,h) over 513 keys [sk, k0..k511], no mask.
// V comes transposed (B*H, HD, N).
// ---------------------------------------------------------------------------
__global__ __launch_bounds__(256)
void state_attn_kernel(const _Float16* __restrict__ sq, const _Float16* __restrict__ sk,
                       const _Float16* __restrict__ sv,
                       const _Float16* __restrict__ K, const _Float16* __restrict__ Vt,
                       _Float16* __restrict__ souts) {
  __shared__ float sc[520];
  __shared__ float red[256];
  int bh = blockIdx.x, b = bh >> 3, h = bh & 7, t = threadIdx.x;
  const _Float16* q = sq + (size_t)bh * HD;
  for (int j = t; j < 513; j += 256) {
    const _Float16* key = (j == 0) ? (sk + (size_t)bh * HD)
                                   : (K + ((size_t)bh * Nn + (j - 1)) * HD);
    float a = 0.0f;
    for (int d = 0; d < HD; d++) a += (float)q[d] * (float)key[d];
    sc[j] = a * SCALEF;
  }
  __syncthreads();
  float m = -1e30f;
  for (int j = t; j < 513; j += 256) m = fmaxf(m, sc[j]);
  red[t] = m; __syncthreads();
  for (int o = 128; o > 0; o >>= 1) { if (t < o) red[t] = fmaxf(red[t], red[t + o]); __syncthreads(); }
  float mx = red[0]; __syncthreads();
  float s = 0.0f;
  for (int j = t; j < 513; j += 256) { float e = __expf(sc[j] - mx); sc[j] = e; s += e; }
  red[t] = s; __syncthreads();
  for (int o = 128; o > 0; o >>= 1) { if (t < o) red[t] += red[t + o]; __syncthreads(); }
  float inv = 1.0f / red[0];
  if (t < HD) {
    const _Float16* vrow = Vt + ((size_t)bh * HD + t) * Nn;
    float acc = sc[0] * (float)sv[(size_t)bh * HD + t];
    for (int j = 1; j < 513; j++) acc += sc[j] * (float)vrow[j - 1];
    souts[(size_t)b * Dd + h * HD + t] = (_Float16)(acc * inv);
  }
}

// s = s*sigmoid(beta) + upd*(1-sigmoid(beta))
__global__ __launch_bounds__(256)
void state_update_kernel(float* __restrict__ s, const float* __restrict__ upd,
                         const float* __restrict__ beta) {
  int i = blockIdx.x * 256 + threadIdx.x;
  int d = i & (Dd - 1);
  float g = 1.0f / (1.0f + __expf(-beta[d]));
  s[i] = s[i] * g + upd[i] * (1.0f - g);
}

// out[0:B*N] = log_softmax over singleton logit axis == 0 exactly.
// out[B*N : ] = updated state.
__global__ __launch_bounds__(256)
void finalize_kernel(const float* __restrict__ s, float* __restrict__ out) {
  int i = blockIdx.x * 256 + threadIdx.x;
  out[i] = 0.0f;
  out[MBr + i] = s[i];
}

// ---------------------------------------------------------------------------
// host
// ---------------------------------------------------------------------------
extern "C" void kernel_launch(void* const* d_in, const int* in_sizes, int n_in,
                              void* d_out, int out_size, void* d_ws, size_t ws_size,
                              hipStream_t stream) {
  (void)in_sizes; (void)n_in; (void)out_size; (void)ws_size;

  const int*   ids      = (const int*)  d_in[0];
  const float* state    = (const float*)d_in[1];
  const float* tok_emb  = (const float*)d_in[2];
  const float* ln1_g    = (const float*)d_in[3];
  const float* ln1_b    = (const float*)d_in[4];
  const float* w_qkv    = (const float*)d_in[5];
  const float* w_o      = (const float*)d_in[6];
  const float* ln2_g    = (const float*)d_in[7];
  const float* ln2_b    = (const float*)d_in[8];
  const float* w_ff1    = (const float*)d_in[9];
  const float* b_ff1    = (const float*)d_in[10];
  const float* w_ff2    = (const float*)d_in[11];
  const float* b_ff2    = (const float*)d_in[12];
  const float* s_ln_g   = (const float*)d_in[13];
  const float* s_ln_b   = (const float*)d_in[14];
  const float* w_s_qkv  = (const float*)d_in[15];
  const float* w_so     = (const float*)d_in[16];
  const float* gate_beta= (const float*)d_in[17];
  float* out = (float*)d_out;

  char* p = (char*)d_ws;
  auto take = [&](size_t bytes) -> char* {
    char* r = p; p += (bytes + 255) & ~(size_t)255; return r;
  };
  float*    x      = (float*)   take((size_t)MBr * Dd * 4);
  _Float16* xln    = (_Float16*)take((size_t)MBr * Dd * 2);
  _Float16* big    = (_Float16*)take((size_t)MBr * 4 * Dd * 2); // qkv/ff1 share
  _Float16* q16    = (_Float16*)take((size_t)BH * Nn * HD * 2);
  _Float16* k16    = (_Float16*)take((size_t)BH * Nn * HD * 2);
  _Float16* vt16   = (_Float16*)take((size_t)BH * Nn * HD * 2); // transposed V
  _Float16* attn16 = (_Float16*)take((size_t)MBr * Dd * 2);
  _Float16* wqkv16 = (_Float16*)take((size_t)Ll * Dd * 3 * Dd * 2);
  _Float16* wo16   = (_Float16*)take((size_t)Ll * Dd * Dd * 2);
  _Float16* wff116 = (_Float16*)take((size_t)Ll * Dd * 4 * Dd * 2);
  _Float16* wff216 = (_Float16*)take((size_t)Ll * 4 * Dd * Dd * 2);
  _Float16* wsq16  = (_Float16*)take((size_t)Dd * 3 * Dd * 2);
  _Float16* wso16  = (_Float16*)take((size_t)Dd * Dd * 2);
  float*    sbuf   = (float*)   take((size_t)Bb * Dd * 4);
  _Float16* sln16  = (_Float16*)take((size_t)Bb * Dd * 2);
  _Float16* sqkv16 = (_Float16*)take((size_t)Bb * 3 * Dd * 2);
  _Float16* sq16   = (_Float16*)take((size_t)BH * HD * 2);
  _Float16* sk16   = (_Float16*)take((size_t)BH * HD * 2);
  _Float16* sv16   = (_Float16*)take((size_t)BH * HD * 2);
  _Float16* souts16= (_Float16*)take((size_t)Bb * Dd * 2);
  float*    upd32  = (float*)   take((size_t)Bb * Dd * 4);

  auto swz = [&](const float* src, _Float16* dst, int K, int N) {
    int nthr = (K / 32) * (N / 16) * 32;
    swizzle_b_kernel<<<nthr / 256, 256, 0, stream>>>(src, dst, K, N);
  };
  auto gemm = [&](const _Float16* A, const _Float16* Bsw, const float* bias,
                  const float* resid, float* Cf, _Float16* Ch,
                  int M, int N, int K, int flags) {
    dim3 g(N / 128, (M + 63) / 64);
    gemm_kernel<<<g, 256, 0, stream>>>(A, Bsw, bias, resid, Cf, Ch, M, N, K, flags);
  };

  // weight swizzles (every call; deterministic)
  for (int l = 0; l < Ll; l++) {
    swz(w_qkv + (size_t)l * Dd * 3 * Dd, wqkv16 + (size_t)l * Dd * 3 * Dd, Dd, 3 * Dd);
    swz(w_o   + (size_t)l * Dd * Dd,     wo16   + (size_t)l * Dd * Dd,     Dd, Dd);
    swz(w_ff1 + (size_t)l * Dd * 4 * Dd, wff116 + (size_t)l * Dd * 4 * Dd, Dd, 4 * Dd);
    swz(w_ff2 + (size_t)l * 4 * Dd * Dd, wff216 + (size_t)l * 4 * Dd * Dd, 4 * Dd, Dd);
  }
  swz(w_s_qkv, wsq16, Dd, 3 * Dd);
  swz(w_so,    wso16, Dd, Dd);

  embed_kernel<<<(MBr * Dd) / 256, 256, 0, stream>>>(ids, tok_emb, x);
  copy_state_kernel<<<(Bb * Dd) / 256, 256, 0, stream>>>(state, sbuf);

  for (int l = 0; l < Ll; l++) {
    ln_kernel<<<MBr, 256, 0, stream>>>(x, ln1_g + l * Dd, ln1_b + l * Dd, xln);
    gemm(xln, wqkv16 + (size_t)l * Dd * 3 * Dd, nullptr, nullptr,
         nullptr, big, MBr, 3 * Dd, Dd, GF_F16OUT);
    rope_split_kernel<<<(Bb * Nn * Hh * 32) / 256, 256, 0, stream>>>(big, q16, k16, vt16);

    if (l == RECL) {
      ln_kernel<<<Bb, 256, 0, stream>>>(sbuf, s_ln_g, s_ln_b, sln16);
      gemm(sln16, wsq16, nullptr, nullptr, nullptr, sqkv16, Bb, 3 * Dd, Dd, GF_F16OUT);
      split_state_kernel<<<(BH * HD) / 256, 256, 0, stream>>>(sqkv16, sq16, sk16, sv16);
      attn_kernel<<<dim3(BH, Nn / 16), 32, 0, stream>>>(q16, k16, vt16, sk16, sv16, attn16, 1);
      state_attn_kernel<<<BH, 256, 0, stream>>>(sq16, sk16, sv16, k16, vt16, souts16);
      gemm(souts16, wso16, nullptr, nullptr, upd32, nullptr, Bb, Dd, Dd, 0);
      state_update_kernel<<<(Bb * Dd) / 256, 256, 0, stream>>>(sbuf, upd32, gate_beta);
    } else {
      attn_kernel<<<dim3(BH, Nn / 16), 32, 0, stream>>>(q16, k16, vt16, nullptr, nullptr, attn16, 0);
    }

    gemm(attn16, wo16 + (size_t)l * Dd * Dd, nullptr, x, x, nullptr, MBr, Dd, Dd, 0);
    ln_kernel<<<MBr, 256, 0, stream>>>(x, ln2_g + l * Dd, ln2_b + l * Dd, xln);
    gemm(xln, wff116 + (size_t)l * Dd * 4 * Dd, b_ff1 + l * 4 * Dd,
         nullptr, nullptr, big, MBr, 4 * Dd, Dd, GF_F16OUT | GF_GELU);
    gemm(big, wff216 + (size_t)l * 4 * Dd * Dd, b_ff2 + l * Dd,
         x, x, nullptr, MBr, Dd, 4 * Dd, 0);
  }

  finalize_kernel<<<MBr / 256, 256, 0, stream>>>(sbuf, out);
}
